// NeighborConsistencyLoss_50843822850471
// MI455X (gfx1250) — matrix operand order repaired and checked by reading, not compiled, not run on hardware
//
#include <hip/hip_runtime.h>
#include <hip/hip_bf16.h>

// ---------- types for WMMA ----------
typedef __attribute__((ext_vector_type(16))) __bf16 v16bf;
typedef __attribute__((ext_vector_type(8)))  __bf16 v8bf;
typedef __attribute__((ext_vector_type(8)))  float  v8f;

#define NROWS 8192
#define DDIM  512
#define NCLS  1000
#define CPAD  1024
#define EPSV  1e-8f

static __device__ __forceinline__ unsigned short f2bf(float x) {
    union { float f; unsigned u; } v; v.f = x;
    unsigned r = v.u + 0x7FFFu + ((v.u >> 16) & 1u);   // round-to-nearest-even
    return (unsigned short)(r >> 16);
}

static __device__ __forceinline__ v16bf cat16(v8bf a, v8bf b) {
    return __builtin_shufflevector(a, b, 0,1,2,3,4,5,6,7,8,9,10,11,12,13,14,15);
}

// ------------------------------------------------------------------
// Kernel 1: softmax rows of logits -> p_f32 [N][CPAD] and transposed
// bf16 p_t [CPAD][N] (pad classes zeroed so WMMA pads contribute 0).
// ------------------------------------------------------------------
__global__ void ncl_softmax_kernel(const float* __restrict__ logits,
                                   float* __restrict__ p32,
                                   unsigned short* __restrict__ pt) {
    const int row = blockIdx.x;
    const int tid = threadIdx.x;
    __shared__ float red[256];

    // pass 1: max
    float m = -3.4e38f;
    for (int c = tid; c < NCLS; c += 256) m = fmaxf(m, logits[row * NCLS + c]);
    red[tid] = m; __syncthreads();
    for (int s = 128; s > 0; s >>= 1) {
        if (tid < s) red[tid] = fmaxf(red[tid], red[tid + s]);
        __syncthreads();
    }
    const float rmax = red[0]; __syncthreads();

    // pass 2: exp + sum (stash exp in p32)
    float acc = 0.f;
    for (int c = tid; c < NCLS; c += 256) {
        float e = __expf(logits[row * NCLS + c] - rmax);
        p32[row * CPAD + c] = e;
        acc += e;
    }
    red[tid] = acc; __syncthreads();
    for (int s = 128; s > 0; s >>= 1) {
        if (tid < s) red[tid] += red[tid + s];
        __syncthreads();
    }
    const float inv = 1.0f / red[0];

    // pass 3: normalize (+eps), write f32 row-major + bf16 transposed
    for (int c = tid; c < CPAD; c += 256) {
        if (c < NCLS) {
            float v = p32[row * CPAD + c] * inv + EPSV;
            p32[row * CPAD + c] = v;
            pt[c * NROWS + row] = f2bf(v);
        } else {
            p32[row * CPAD + c] = 0.f;
            pt[c * NROWS + row] = 0;
        }
    }
}

// ------------------------------------------------------------------
// Kernel 2: features f32 -> bf16, and zero the scalar output
// ------------------------------------------------------------------
__global__ void ncl_cvt_kernel(const float* __restrict__ f,
                               unsigned short* __restrict__ fb,
                               float* __restrict__ out) {
    int i = (blockIdx.x * blockDim.x + threadIdx.x) * 4;
    const int n = NROWS * DDIM;
    if (i == 0) out[0] = 0.f;
    if (i < n) {
        fb[i + 0] = f2bf(f[i + 0]);
        fb[i + 1] = f2bf(f[i + 1]);
        fb[i + 2] = f2bf(f[i + 2]);
        fb[i + 3] = f2bf(f[i + 3]);
    }
}

// ------------------------------------------------------------------
// Kernel 3: fused  S = F·F^T - I ;  w = mask(S)/rowsum ;  out = w@p ;
//           kl += out*(log out - log p)  -- one pass, S never hits HBM.
// Block = 256 thr (8 waves), owns 16 output rows.
//   wave w: builds S tile cols [j0+16w, j0+16w+16) each iter,
//           owns output cols [128w, 128w+128) as 8 f32 accum tiles.
// ------------------------------------------------------------------
__global__ void __launch_bounds__(256)
ncl_fused_kernel(const unsigned short* __restrict__ fbf,   // [N][D] bf16
                 const unsigned short* __restrict__ pt,    // [CPAD][N] bf16
                 const float* __restrict__ p32,            // [N][CPAD] f32
                 float* __restrict__ dout) {
    __shared__ unsigned short sF[16 * DDIM];   // F rows of this block (16 KB)
    __shared__ unsigned short sS[16 * 128];    // masked S chunk, bf16 (4 KB)
    __shared__ float s_rowsum[16];
    __shared__ float s_kl[8];

    const int tid  = threadIdx.x;
    const int wid  = tid >> 5;          // wave id 0..7
    const int lane = tid & 31;
    const int half = lane >> 4;         // 0 | 1
    const int ln   = lane & 15;
    const int I0   = blockIdx.x * 16;   // global row base
    const int cb   = wid * 128;         // this wave's class-column base

    // stage F_I into LDS (16 x 512 bf16), 16-ushort chunks per thread
    for (int ch = tid; ch < (16 * DDIM) / 16; ch += 256) {
        *(uint4*)&sF[ch * 16 + 0] = *(const uint4*)&fbf[I0 * DDIM + ch * 16 + 0];
        *(uint4*)&sF[ch * 16 + 8] = *(const uint4*)&fbf[I0 * DDIM + ch * 16 + 8];
    }
    if (tid < 16) s_rowsum[tid] = 0.f;

    v8f acc[8];
#pragma unroll
    for (int t = 0; t < 8; ++t) acc[t] = (v8f)(0.f);
    float lane_sum[8] = {0.f, 0.f, 0.f, 0.f, 0.f, 0.f, 0.f, 0.f};

    __syncthreads();

    for (int j0 = 0; j0 < NROWS; j0 += 128) {
        // ---- GEMM1: S tile (16 x 16) = F_I (16xK) x F_J^T (Kx16), K=512 ----
        const int jcol = j0 + wid * 16 + ln;               // S column / F row
        const unsigned short* Bp = fbf + jcol * DDIM + half * 16;
        if (jcol + 128 < NROWS)
            __builtin_prefetch((const void*)(Bp + 128 * DDIM), 0, 1);

        v8f s = (v8f)(0.f);
#pragma unroll
        for (int k = 0; k < DDIM; k += 32) {
            v16bf a = *(const v16bf*)&sF[ln * DDIM + k + half * 16];
            v16bf b = *(const v16bf*)&Bp[k];
            s = __builtin_amdgcn_wmma_f32_16x16x32_bf16(
                    false, a, false, b, (short)0, s, false, false);
        }

        // ---- diag subtract, neighbor mask, rowsum, bf16 -> LDS ----
#pragma unroll
        for (int g = 0; g < 8; ++g) {
            const int rloc = g + half * 8;                 // local row 0..15
            float v = s[g] - ((I0 + rloc) == jcol ? 1.0f : 0.0f);
            v = (v >= -1.0f) ? v : 0.0f;                   // mask
            lane_sum[g] += v;
            sS[rloc * 128 + wid * 16 + ln] = f2bf(v);
        }
        __syncthreads();

        // ---- GEMM2: acc(16 x 128) += S_chunk(16x128) x p_t-block(128x128) ----
#pragma unroll
        for (int kk = 0; kk < 128; kk += 32) {
            v16bf a = *(const v16bf*)&sS[ln * 128 + kk + half * 16];
#pragma unroll
            for (int t = 0; t < 8; ++t) {
                const int ct = cb + t * 16 + ln;           // class column
                v16bf b = *(const v16bf*)&pt[ct * NROWS + j0 + kk + half * 16];
                acc[t] = __builtin_amdgcn_wmma_f32_16x16x32_bf16(
                             false, a, false, b, (short)0, acc[t], false, false);
            }
        }
        __syncthreads();   // protect sS for next iteration
    }

    // ---- finalize rowsum: reduce 16 lanes sharing a row, LDS atomic ----
#pragma unroll
    for (int g = 0; g < 8; ++g) {
        float r = lane_sum[g];
        for (int m = 1; m < 16; m <<= 1) r += __shfl_xor(r, m, 16);
        if (ln == 0) atomicAdd(&s_rowsum[g + half * 8], r);
    }
    __syncthreads();

    // ---- divide, KL contribution ----
    float kl = 0.f;
#pragma unroll
    for (int g = 0; g < 8; ++g) {
        const int rloc = g + half * 8;
        const float inv = 1.0f / s_rowsum[rloc];
#pragma unroll
        for (int t = 0; t < 8; ++t) {
            const int col = cb + t * 16 + ln;
            if (col < NCLS) {
                float o  = acc[t][g] * inv;
                float pv = p32[(I0 + rloc) * CPAD + col];
                kl += o * (__logf(o) - __logf(pv));
            }
        }
    }
    for (int m = 16; m >= 1; m >>= 1) kl += __shfl_xor(kl, m, 32);
    if (lane == 0) s_kl[wid] = kl;
    __syncthreads();
    if (tid == 0) {
        float tot = 0.f;
#pragma unroll
        for (int w = 0; w < 8; ++w) tot += s_kl[w];
        atomicAdd(dout, tot * (1.0f / (float)NROWS));
    }
}

// ------------------------------------------------------------------
extern "C" void kernel_launch(void* const* d_in, const int* in_sizes, int n_in,
                              void* d_out, int out_size, void* d_ws, size_t ws_size,
                              hipStream_t stream) {
    const float* features = (const float*)d_in[0];   // [8192*512]
    const float* logits   = (const float*)d_in[1];   // [8192*1000]
    float* out = (float*)d_out;

    // workspace layout (total 56 MB):
    unsigned short* fbf = (unsigned short*)d_ws;                              // 8 MB
    float*          p32 = (float*)((char*)d_ws + (size_t)NROWS * DDIM * 2);   // 32 MB
    unsigned short* pt  = (unsigned short*)((char*)p32 + (size_t)NROWS * CPAD * 4); // 16 MB

    ncl_softmax_kernel<<<NROWS, 256, 0, stream>>>(logits, p32, pt);
    ncl_cvt_kernel<<<(NROWS * DDIM / 4 + 255) / 256, 256, 0, stream>>>(features, fbf, out);
    ncl_fused_kernel<<<NROWS / 16, 256, 0, stream>>>(fbf, pt, p32, out);
}
